// Model_37194416783889
// MI455X (gfx1250) — matrix operand order repaired and checked by reading, not compiled, not run on hardware
//
#include <hip/hip_runtime.h>

// ---------------------------------------------------------------------------
// Model dims
#define B_    4096
#define IN_   36
#define L_    36
#define D_    128
#define DFF_  512
#define EL_   8
#define LD_   (L_ * D_)        // 4608
#define FC1N_ 13824
#define QKVN_ 384
// ---------------------------------------------------------------------------

typedef __attribute__((ext_vector_type(16))) __bf16 v16bf;
typedef __attribute__((ext_vector_type(8)))  __bf16 v8bf;
typedef __attribute__((ext_vector_type(8)))  float  v8f;

__device__ __forceinline__ float gelu_exact(float x) {
  return 0.5f * x * (1.0f + erff(x * 0.70710678118654752f));
}

// ---------------------------------------------------------------------------
// CDNA5 async global->LDS staging (ASYNCcnt path), with sync fallback.
#if defined(__has_builtin)
#  if __has_builtin(__builtin_amdgcn_global_load_async_to_lds_b128)
#    define HAVE_ASYNC_LDS 1
#  endif
#endif
#ifndef HAVE_ASYNC_LDS
#  define HAVE_ASYNC_LDS 0
#endif

#if HAVE_ASYNC_LDS
typedef int vi4 __attribute__((vector_size(16)));        // matches builtin pointee
typedef __attribute__((address_space(1))) vi4 as1_vi4;   // global (printed __device__)
typedef __attribute__((address_space(3))) vi4 as3_vi4;   // LDS
__device__ __forceinline__ void async_b128(const float* g, const float* lds_dst) {
  // generic->AS1 is numerically identity; generic LDS addr[31:0] == LDS offset.
  __builtin_amdgcn_global_load_async_to_lds_b128(
      (as1_vi4*)(unsigned long long)(size_t)g,
      (as3_vi4*)(unsigned int)(size_t)lds_dst, 0, 0);
}
__device__ __forceinline__ void async_wait0() {
#  if __has_builtin(__builtin_amdgcn_s_wait_asynccnt)
  __builtin_amdgcn_s_wait_asynccnt(0);
#  else
  asm volatile("s_wait_asynccnt 0" ::: "memory");
#  endif
}
#endif

// ---------------------------------------------------------------------------
// WMMA GEMM:  C[M,N] = act( A[M,K](bf16) * BT[N,K](bf16)^T + bias (+ res) )
// Block: 256 threads = 8 waves; block tile 128(M) x 128(N); wave tile 32x64.
// Requires M%128==0, N%128==0, K%32==0 (guaranteed by launcher).
// ---------------------------------------------------------------------------
__global__ __launch_bounds__(256)
void k_gemm_bf16(const __bf16* __restrict__ A, const __bf16* __restrict__ BT,
                 const float* __restrict__ bias, const float* __restrict__ res,
                 float* __restrict__ Cf, __bf16* __restrict__ Cb,
                 int M, int N, int K, int actGelu) {
  const int lane = threadIdx.x & 31;
  const int wave = threadIdx.x >> 5;
  const int m0 = blockIdx.y * 128 + (wave & 3) * 32;
  const int n0 = blockIdx.x * 128 + (wave >> 2) * 64;
  const int half = lane >> 4;      // 0: lanes 0-15, 1: lanes 16-31
  const int l15  = lane & 15;

  v8f acc[2][4] = {};

  const __bf16* arow0 = A + (size_t)(m0 + l15) * K;
  const __bf16* arow1 = arow0 + (size_t)16 * K;
  const int aoff = half ? 8 : 0;   // A: upper lanes hold K 8..15 / 24..31
  const int boff = half ? 8 : 0;   // B: upper lanes hold K 8..15 / 24..31

  for (int k0 = 0; k0 < K; k0 += 32) {
    v8bf a0lo = *(const v8bf*)(arow0 + k0 + aoff);
    v8bf a0hi = *(const v8bf*)(arow0 + k0 + aoff + 16);
    v8bf a1lo = *(const v8bf*)(arow1 + k0 + aoff);
    v8bf a1hi = *(const v8bf*)(arow1 + k0 + aoff + 16);
    v16bf a0 = __builtin_shufflevector(a0lo, a0hi, 0,1,2,3,4,5,6,7,8,9,10,11,12,13,14,15);
    v16bf a1 = __builtin_shufflevector(a1lo, a1hi, 0,1,2,3,4,5,6,7,8,9,10,11,12,13,14,15);
    __builtin_prefetch(arow0 + k0 + 32, 0, 1);   // global_prefetch_b8
    __builtin_prefetch(arow1 + k0 + 32, 0, 1);
#pragma unroll
    for (int j = 0; j < 4; ++j) {
      const __bf16* bp = BT + (size_t)(n0 + j * 16 + l15) * K + k0 + boff;
      v8bf blo = *(const v8bf*)(bp);
      v8bf bhi = *(const v8bf*)(bp + 16);
      v16bf bv = __builtin_shufflevector(blo, bhi, 0,1,2,3,4,5,6,7,8,9,10,11,12,13,14,15);
      acc[0][j] = __builtin_amdgcn_wmma_f32_16x16x32_bf16(
          false, a0, false, bv, (short)0, acc[0][j], false, false);
      acc[1][j] = __builtin_amdgcn_wmma_f32_16x16x32_bf16(
          false, a1, false, bv, (short)0, acc[1][j], false, false);
    }
  }

#pragma unroll
  for (int mi = 0; mi < 2; ++mi) {
    const int rbase = m0 + mi * 16 + half * 8;
#pragma unroll
    for (int j = 0; j < 4; ++j) {
      const int col = n0 + j * 16 + l15;
      const float bvv = bias ? bias[col] : 0.0f;
#pragma unroll
      for (int r = 0; r < 8; ++r) {
        const size_t idx = (size_t)(rbase + r) * N + col;
        float v = acc[mi][j][r] + bvv;
        if (res) v += res[idx];
        if (actGelu) v = gelu_exact(v);
        if (Cf) Cf[idx] = v;
        if (Cb) Cb[idx] = (__bf16)v;
      }
    }
  }
}

// ---------------------------------------------------------------------------
// src [K,N] f32 row-major -> dst [N,Kp] bf16 (transpose, zero-pad K..Kp)
__global__ void k_cvtT(const float* __restrict__ s, __bf16* __restrict__ d,
                       int K, int N, int Kp) {
  size_t i = (size_t)blockIdx.x * 256 + threadIdx.x;
  if (i >= (size_t)N * Kp) return;
  int kp = (int)(i % Kp);
  int n  = (int)(i / Kp);
  d[i] = (kp < K) ? (__bf16)s[(size_t)kp * N + n] : (__bf16)0.0f;
}

// src [M,K] f32 -> dst [M,Kp] bf16 (row zero-pad)
__global__ void k_cvtPad(const float* __restrict__ s, __bf16* __restrict__ d,
                         int K, int Kp, size_t M) {
  size_t i = (size_t)blockIdx.x * 256 + threadIdx.x;
  if (i >= M * (size_t)Kp) return;
  int kp = (int)(i % Kp);
  size_t m = i / Kp;
  d[i] = (kp < K) ? (__bf16)s[m * K + kp] : (__bf16)0.0f;
}

// concat [bq|bk|bv] per layer -> qkvb [EL,384]
__global__ void k_bias3(const float* __restrict__ bq, const float* __restrict__ bk,
                        const float* __restrict__ bv, float* __restrict__ dst) {
  int i = blockIdx.x * 256 + threadIdx.x;
  if (i >= EL_ * QKVN_) return;
  int l = i / QKVN_, j = i % QKVN_;
  float v = (j < 128) ? bq[l * 128 + j]
          : (j < 256) ? bk[l * 128 + (j - 128)]
                      : bv[l * 128 + (j - 256)];
  dst[i] = v;
}

// ---------------------------------------------------------------------------
// Autocorrelation (one block per batch element b). qkv: [B*L, 384] fused rows.
//   corr[tau] = mean_d sum_t q[(t+tau)%L,d]*k[t,d]  == irfft(rfft(q)conj(rfft(k)))
//   top-3 taus, softmax weights, out[t,d] = sum_i w_i * v[(t+delay_i)%L, d]
__global__ __launch_bounds__(256)
void k_autocorr(const float* __restrict__ qkv, __bf16* __restrict__ out) {
  __shared__ float sq[LD_], sk[LD_], sv[LD_];
  __shared__ float corr[L_];
  __shared__ float w3[3];
  __shared__ int   d3[3];
  const size_t rowbase = (size_t)blockIdx.x * L_ * QKVN_;
#if HAVE_ASYNC_LDS
  for (int i = threadIdx.x * 4; i < LD_; i += 1024) {
    const int t = i >> 7, d = i & 127;
    const float* rp = qkv + rowbase + (size_t)t * QKVN_ + d;
    async_b128(rp,       &sq[i]);
    async_b128(rp + 128, &sk[i]);
    async_b128(rp + 256, &sv[i]);
  }
  async_wait0();
#else
  for (int i = threadIdx.x; i < LD_; i += 256) {
    const int t = i >> 7, d = i & 127;
    const float* rp = qkv + rowbase + (size_t)t * QKVN_ + d;
    sq[i] = rp[0]; sk[i] = rp[128]; sv[i] = rp[256];
  }
#endif
  if (threadIdx.x < L_) corr[threadIdx.x] = 0.0f;
  __syncthreads();
  for (int i = threadIdx.x; i < LD_; i += 256) {
    const int tau = i >> 7, d = i & 127;
    float s = 0.0f;
    for (int t = 0; t < L_; ++t) {
      int tp = t + tau; if (tp >= L_) tp -= L_;
      s += sq[tp * D_ + d] * sk[t * D_ + d];
    }
    atomicAdd(&corr[tau], s);
  }
  __syncthreads();
  if (threadIdx.x == 0) {
    float b0 = -1e30f, b1 = -1e30f, b2 = -1e30f;
    int i0 = 0, i1 = 0, i2 = 0;
    for (int tau = 0; tau < L_; ++tau) {
      float val = corr[tau] * (1.0f / (float)D_);   // mean over H*E = 128 channels
      if (val > b0)      { b2 = b1; i2 = i1; b1 = b0; i1 = i0; b0 = val; i0 = tau; }
      else if (val > b1) { b2 = b1; i2 = i1; b1 = val; i1 = tau; }
      else if (val > b2) { b2 = val; i2 = tau; }
    }
    float e1 = expf(b1 - b0), e2 = expf(b2 - b0);
    float inv = 1.0f / (1.0f + e1 + e2);
    w3[0] = inv; w3[1] = e1 * inv; w3[2] = e2 * inv;
    d3[0] = i0; d3[1] = i1; d3[2] = i2;
  }
  __syncthreads();
  const size_t obase = (size_t)blockIdx.x * LD_;
  for (int i = threadIdx.x; i < LD_; i += 256) {
    const int t = i >> 7, d = i & 127;
    float s = 0.0f;
#pragma unroll
    for (int j = 0; j < 3; ++j) {
      int tp = t + d3[j]; if (tp >= L_) tp -= L_;
      s += w3[j] * sv[tp * D_ + d];
    }
    out[obase + i] = (__bf16)s;
  }
}

// ---------------------------------------------------------------------------
// Series decomposition: h = h - movavg_k25_edgepad(h); writes f32 (in place)
// and bf16 copy for the next GEMM. One block per batch element.
__global__ __launch_bounds__(256)
void k_decomp(float* __restrict__ h, __bf16* __restrict__ hb) {
  __shared__ float sh[LD_];
  const size_t base = (size_t)blockIdx.x * LD_;
#if HAVE_ASYNC_LDS
  for (int i = threadIdx.x * 4; i < LD_; i += 1024) async_b128(h + base + i, &sh[i]);
  async_wait0();
#else
  for (int i = threadIdx.x; i < LD_; i += 256) sh[i] = h[base + i];
#endif
  __syncthreads();
  for (int i = threadIdx.x; i < LD_; i += 256) {
    const int t = i >> 7, d = i & 127;
    float s = 0.0f;
#pragma unroll
    for (int j = -12; j <= 12; ++j) {
      int tt = t + j;
      tt = tt < 0 ? 0 : (tt > L_ - 1 ? L_ - 1 : tt);
      s += sh[tt * D_ + d];
    }
    const float v = sh[i] - s * (1.0f / 25.0f);
    h[base + i] = v;
    hb[base + i] = (__bf16)v;
  }
}

// ---------------------------------------------------------------------------
// my_Layernorm: LN over features, subtract per-(b,d) time-mean; bf16 out only.
__global__ __launch_bounds__(256)
void k_ln(const float* __restrict__ h, const float* __restrict__ w,
          const float* __restrict__ bta, __bf16* __restrict__ out) {
  __shared__ float sh[LD_];
  __shared__ float xh[LD_];
  __shared__ float cm[D_];
  const size_t base = (size_t)blockIdx.x * LD_;
#if HAVE_ASYNC_LDS
  for (int i = threadIdx.x * 4; i < LD_; i += 1024) async_b128(h + base + i, &sh[i]);
  async_wait0();
#else
  for (int i = threadIdx.x; i < LD_; i += 256) sh[i] = h[base + i];
#endif
  __syncthreads();
  for (int t = threadIdx.x; t < L_; t += 256) {
    float mu = 0.0f;
    for (int d = 0; d < D_; ++d) mu += sh[t * D_ + d];
    mu *= (1.0f / (float)D_);
    float var = 0.0f;
    for (int d = 0; d < D_; ++d) { float df = sh[t * D_ + d] - mu; var += df * df; }
    var *= (1.0f / (float)D_);
    const float inv = rsqrtf(var + 1e-5f);
    for (int d = 0; d < D_; ++d)
      xh[t * D_ + d] = (sh[t * D_ + d] - mu) * inv * w[d] + bta[d];
  }
  __syncthreads();
  for (int d = threadIdx.x; d < D_; d += 256) {
    float s = 0.0f;
    for (int t = 0; t < L_; ++t) s += xh[t * D_ + d];
    cm[d] = s * (1.0f / (float)L_);
  }
  __syncthreads();
  for (int i = threadIdx.x; i < LD_; i += 256)
    out[base + i] = (__bf16)(xh[i] - cm[i & 127]);
}

// ---------------------------------------------------------------------------
// fc3 (N=2) + softmax. One wave per batch row, coalesced K, shfl reduction.
__global__ __launch_bounds__(256)
void k_head(const float* __restrict__ z, const float* __restrict__ w,
            const float* __restrict__ b, float* __restrict__ out) {
  const int wave = threadIdx.x >> 5, lane = threadIdx.x & 31;
  const int row = blockIdx.x * 8 + wave;
  const float* zr = z + (size_t)row * LD_;
  float a0 = 0.0f, a1 = 0.0f;
  for (int kk = lane; kk < LD_; kk += 32) {
    const float x = zr[kk];
    a0 += x * w[kk * 2];
    a1 += x * w[kk * 2 + 1];
  }
#pragma unroll
  for (int m = 16; m >= 1; m >>= 1) {
    a0 += __shfl_xor(a0, m, 32);
    a1 += __shfl_xor(a1, m, 32);
  }
  if (lane == 0) {
    a0 += b[0]; a1 += b[1];
    const float mx = fmaxf(a0, a1);
    const float e0 = expf(a0 - mx), e1 = expf(a1 - mx);
    const float inv = 1.0f / (e0 + e1);
    out[row * 2]     = e0 * inv;
    out[row * 2 + 1] = e1 * inv;
  }
}

// ---------------------------------------------------------------------------
extern "C" void kernel_launch(void* const* d_in, const int* in_sizes, int n_in,
                              void* d_out, int out_size, void* d_ws, size_t ws_size,
                              hipStream_t stream) {
  (void)in_sizes; (void)n_in; (void)out_size; (void)ws_size;
  const float* x        = (const float*)d_in[0];
  const float* linear_w = (const float*)d_in[1];
  const float* linear_b = (const float*)d_in[2];
  const float* Wq = (const float*)d_in[3];
  const float* bq = (const float*)d_in[4];
  const float* Wk = (const float*)d_in[5];
  const float* bk = (const float*)d_in[6];
  const float* Wv = (const float*)d_in[7];
  const float* bv = (const float*)d_in[8];
  const float* Wo = (const float*)d_in[9];
  const float* bo = (const float*)d_in[10];
  const float* conv1_w = (const float*)d_in[11];
  const float* conv2_w = (const float*)d_in[12];
  const float* ln_w = (const float*)d_in[13];
  const float* ln_b = (const float*)d_in[14];
  const float* fc1_w = (const float*)d_in[15];
  const float* fc1_b = (const float*)d_in[16];
  const float* fc2_w = (const float*)d_in[17];
  const float* fc2_b = (const float*)d_in[18];
  const float* fc3_w = (const float*)d_in[19];
  const float* fc3_b = (const float*)d_in[20];

  char* ws = (char*)d_ws;
  size_t off = 0;
  auto alloc = [&](size_t bytes) -> void* {
    void* p = ws + off;
    off += (bytes + 255) & ~(size_t)255;
    return p;
  };

  const size_t M1 = (size_t)B_ * L_;                 // 147456 rows
  float*  h    = (float*)alloc(M1 * D_ * 4);         // [M1,128] / [4096,4608]
  float*  qkv  = (float*)alloc(M1 * QKVN_ * 4);      // fused q|k|v
  float*  tb   = (float*)alloc(M1 * D_ * 4);         // fc2 output [4096,4608]
  __bf16* abf  = (__bf16*)alloc(M1 * D_ * 2);        // bf16 activations [M1,128]
  __bf16* bbf  = (__bf16*)alloc(M1 * DFF_ * 2);      // bf16 FFN/fc1 activations
  __bf16* xpad = (__bf16*)alloc((size_t)B_ * 64 * 2);
  __bf16* linT = (__bf16*)alloc((size_t)LD_ * 64 * 2);
  __bf16* qkvT = (__bf16*)alloc((size_t)EL_ * QKVN_ * D_ * 2);
  __bf16* woT  = (__bf16*)alloc((size_t)EL_ * D_ * D_ * 2);
  __bf16* c1T  = (__bf16*)alloc((size_t)EL_ * DFF_ * D_ * 2);
  __bf16* c2T  = (__bf16*)alloc((size_t)EL_ * D_ * DFF_ * 2);
  __bf16* f1T  = (__bf16*)alloc((size_t)FC1N_ * LD_ * 2);
  __bf16* f2T  = (__bf16*)alloc((size_t)LD_ * FC1N_ * 2);
  float*  qkvb = (float*)alloc((size_t)EL_ * QKVN_ * 4);

  auto cgrid = [](size_t n) { return dim3((unsigned)((n + 255) / 256)); };
  auto gemm = [&](const __bf16* A, const __bf16* BT, const float* bias,
                  const float* res, float* Cf, __bf16* Cb,
                  int M, int N, int K, int act) {
    dim3 g(N / 128, M / 128);
    k_gemm_bf16<<<g, 256, 0, stream>>>(A, BT, bias, res, Cf, Cb, M, N, K, act);
  };

  // ---- weight prep (deterministic, every call) ----
  k_cvtPad<<<cgrid((size_t)B_ * 64), 256, 0, stream>>>(x, xpad, IN_, 64, B_);
  k_cvtT<<<cgrid((size_t)LD_ * 64), 256, 0, stream>>>(linear_w, linT, IN_, LD_, 64);
  for (int l = 0; l < EL_; ++l) {
    const size_t dd = (size_t)D_ * D_;
    __bf16* qt = qkvT + (size_t)l * QKVN_ * D_;
    k_cvtT<<<cgrid(dd), 256, 0, stream>>>(Wq + l * dd, qt,            D_, D_, D_);
    k_cvtT<<<cgrid(dd), 256, 0, stream>>>(Wk + l * dd, qt + dd,       D_, D_, D_);
    k_cvtT<<<cgrid(dd), 256, 0, stream>>>(Wv + l * dd, qt + 2 * dd,   D_, D_, D_);
    k_cvtT<<<cgrid(dd), 256, 0, stream>>>(Wo + l * dd, woT + l * dd,  D_, D_, D_);
    const size_t df = (size_t)D_ * DFF_;
    k_cvtT<<<cgrid(df), 256, 0, stream>>>(conv1_w + l * df, c1T + l * df, D_, DFF_, D_);
    k_cvtT<<<cgrid(df), 256, 0, stream>>>(conv2_w + l * df, c2T + l * df, DFF_, D_, DFF_);
  }
  k_bias3<<<cgrid((size_t)EL_ * QKVN_), 256, 0, stream>>>(bq, bk, bv, qkvb);
  k_cvtT<<<cgrid((size_t)FC1N_ * LD_), 256, 0, stream>>>(fc1_w, f1T, LD_, FC1N_, LD_);
  k_cvtT<<<cgrid((size_t)LD_ * FC1N_), 256, 0, stream>>>(fc2_w, f2T, FC1N_, LD_, FC1N_);

  // ---- embedding: h = x @ linear_w + linear_b (also bf16 copy) ----
  gemm(xpad, linT, linear_b, nullptr, h, abf, B_, LD_, 64, 0);

  // ---- encoder layers ----
  for (int l = 0; l < EL_; ++l) {
    const size_t dd = (size_t)D_ * D_;
    const size_t df = (size_t)D_ * DFF_;
    // fused q|k|v projection: [M1,384]
    gemm(abf, qkvT + (size_t)l * QKVN_ * D_, qkvb + l * QKVN_, nullptr,
         qkv, nullptr, (int)M1, QKVN_, D_, 0);
    k_autocorr<<<B_, 256, 0, stream>>>(qkv, abf);                 // att -> bf16
    gemm(abf, woT + l * dd, bo + l * D_, h, h, nullptr, (int)M1, D_, D_, 0);
    k_decomp<<<B_, 256, 0, stream>>>(h, abf);                     // decomp 1
    gemm(abf, c1T + l * df, nullptr, nullptr, nullptr, bbf, (int)M1, DFF_, D_, 1);
    gemm(bbf, c2T + l * df, nullptr, h, h, nullptr, (int)M1, D_, DFF_, 0);
    k_decomp<<<B_, 256, 0, stream>>>(h, abf);                     // decomp 2
  }

  // ---- head ----
  k_ln<<<B_, 256, 0, stream>>>(h, ln_w, ln_b, abf);               // -> bf16 [4096,4608]
  gemm(abf, f1T, fc1_b, nullptr, nullptr, bbf, B_, FC1N_, LD_, 1);
  gemm(bbf, f2T, fc2_b, nullptr, tb, nullptr, B_, LD_, FC1N_, 0);
  k_head<<<(B_ / 8), 256, 0, stream>>>(tb, fc3_w, fc3_b, (float*)d_out);
}
